// LSTMImageAlice_6150393167918
// MI455X (gfx1250) — compile-verified, matrix operand order checked by reading, not run on hardware
//
#include <hip/hip_runtime.h>

// ---------------- problem constants (from reference) ----------------
#define Bsz   1024
#define Hdim  512
#define Vocab 8192
#define Tlen  20
#define Kxh   1024   // [x | h] concat width
#define Ngate 2048   // 4*H

// ---------------- WMMA vector types ----------------
typedef __bf16 bf16_t;
typedef bf16_t v16bf __attribute__((ext_vector_type(16)));
typedef float  v8f   __attribute__((ext_vector_type(8)));

// ---------------- helpers ----------------
__device__ __forceinline__ unsigned short f2bf(float x) {
    unsigned int u = __float_as_uint(x);
    unsigned int r = (u + 0x7FFFu + ((u >> 16) & 1u)) >> 16;
    return (unsigned short)r;
}

__device__ __forceinline__ float sigmoidf_(float x) {
    return 1.0f / (1.0f + __expf(-x));
}

__device__ __forceinline__ unsigned long long mix64(unsigned long long z) {
    z += 0x9E3779B97F4A7C15ull;
    z = (z ^ (z >> 30)) * 0xBF58476D1CE4E5B9ull;
    z = (z ^ (z >> 27)) * 0x94D049BB133111EBull;
    return z ^ (z >> 31);
}

// CDNA5 async copy: global -> LDS, 16 bytes per lane, tracked by ASYNCcnt.
// LDS address = low 32 bits of the generic shared pointer (aperture decode).
__device__ __forceinline__ void async_ld_b128(void* lds, const void* gaddr) {
    unsigned       l = (unsigned)(unsigned long long)(uintptr_t)lds;
    unsigned long long g = (unsigned long long)(uintptr_t)gaddr;
    asm volatile("global_load_async_to_lds_b128 %0, %1, off"
                 :: "v"(l), "v"(g)
                 : "memory");
}

#define WAIT_ASYNC_0()  asm volatile("s_wait_asynccnt 0x0" ::: "memory")
#define WAIT_ASYNC_10() asm volatile("s_wait_asynccnt 0xa" ::: "memory")

// ---------------- small utility kernels ----------------
__global__ void cvt_f32_bf16(const float* __restrict__ src,
                             unsigned short* __restrict__ dst, int n) {
    int i = blockIdx.x * blockDim.x + threadIdx.x;
    if (i < n) dst[i] = f2bf(src[i]);
}

// wcat[j][0:512] = w_ih[j], wcat[j][512:1024] = w_hh[j]  (bf16)
__global__ void build_wcat(const float* __restrict__ w_ih,
                           const float* __restrict__ w_hh,
                           unsigned short* __restrict__ wcat) {
    int i = blockIdx.x * blockDim.x + threadIdx.x;   // over 2048*1024
    if (i >= Ngate * Kxh) return;
    int j = i >> 10;            // row
    int k = i & 1023;           // col
    float v = (k < Hdim) ? w_ih[(size_t)j * Hdim + k]
                         : w_hh[(size_t)j * Hdim + (k - Hdim)];
    wcat[i] = f2bf(v);
}

__global__ void build_bcat(const float* __restrict__ b_ih,
                           const float* __restrict__ b_hh,
                           float* __restrict__ bcat) {
    int i = blockIdx.x * blockDim.x + threadIdx.x;
    if (i < Ngate) bcat[i] = b_ih[i] + b_hh[i];
}

__global__ void zero_f32(float* __restrict__ p, int n) {
    int i = blockIdx.x * blockDim.x + threadIdx.x;
    if (i < n) p[i] = 0.0f;
}

__global__ void fill_f32(float* __restrict__ p, float v, int n) {
    int i = blockIdx.x * blockDim.x + threadIdx.x;
    if (i < n) p[i] = v;
}

// zero the x half of xh ([b][0:512])
__global__ void zero_xh_x(unsigned short* __restrict__ xh) {
    int i = blockIdx.x * blockDim.x + threadIdx.x;   // over B*H
    if (i >= Bsz * Hdim) return;
    int b = i >> 9, j = i & 511;
    xh[(size_t)b * Kxh + j] = 0;
}

// ---------------- WMMA GEMM: C[M,N] = A[M,K] @ W[N,K]^T + bias ----------------
// bf16 row-major operands. Block tile 64(M) x 256(N), 8 wave32s (2x4);
// each wave owns a 32x64 sub-tile = 2x4 of 16x16x32 WMMAs.
// K staged through double-buffered LDS filled with GLOBAL_LOAD_ASYNC_TO_LDS_B128
// (ASYNCcnt), overlapping the next chunk's copy with the current chunk's WMMAs.
#define LDS_STRIDE 80                       // elems; 160B rows keep 32B alignment
#define TILE_ELEMS ((64 + 256) * LDS_STRIDE)

__global__ __launch_bounds__(256)
void wmma_gemm_bf16(const unsigned short* __restrict__ A, int lda,
                    const unsigned short* __restrict__ W, int ldw,
                    const float* __restrict__ bias,
                    float* __restrict__ Cf, int ldc,
                    unsigned short* __restrict__ Cb, int ldcb,
                    int K, int relu) {
    __shared__ __attribute__((aligned(32))) unsigned short smem[2 * TILE_ELEMS];

    const int tid  = threadIdx.x;
    const int lane = tid & 31;
    const int wid  = tid >> 5;       // 0..7
    const int wm   = wid & 1;        // wave position in M (2)
    const int wn   = wid >> 1;       // wave position in N (4)
    const int lrow = lane & 15;
    const int lkh  = (lane >> 4) << 4;   // K half select: 0 or 16

    const int blockM = blockIdx.x * 64;
    const int blockN = blockIdx.y * 256;

    // per-thread source pointers / LDS slots for the cooperative fill:
    //   A tile 64x64 : thread -> row tid/4, col-seg (tid%4)*16  (2 x b128)
    //   W tile 256x64: thread -> row tid, cols 0..63            (8 x b128)
    const int ar = tid >> 2;
    const int ac = (tid & 3) << 4;
    const unsigned short* gA = A + (size_t)(blockM + ar) * lda + ac;
    const unsigned short* gW = W + (size_t)(blockN + tid) * ldw;

    v8f acc[2][4] = {};

    const int nk = K >> 6;   // K chunks of 64

    // ---- prologue: async-fill chunk 0 into buffer 0 ----
    {
        unsigned short* bA = smem;
        unsigned short* bW = smem + 64 * LDS_STRIDE;
        async_ld_b128(&bA[ar * LDS_STRIDE + ac],     gA);
        async_ld_b128(&bA[ar * LDS_STRIDE + ac + 8], gA + 8);
#pragma unroll
        for (int j = 0; j < 8; j++)
            async_ld_b128(&bW[tid * LDS_STRIDE + j * 8], gW + j * 8);
    }

    for (int i = 0; i < nk; i++) {
        const int cur = i & 1;
        unsigned short* curA = smem + cur * TILE_ELEMS;
        unsigned short* curW = curA + 64 * LDS_STRIDE;

        if (i + 1 < nk) {
            // issue next chunk into the other buffer, then wait for current
            const int k0n = (i + 1) << 6;
            unsigned short* nA = smem + (cur ^ 1) * TILE_ELEMS;
            unsigned short* nW = nA + 64 * LDS_STRIDE;
            async_ld_b128(&nA[ar * LDS_STRIDE + ac],     gA + k0n);
            async_ld_b128(&nA[ar * LDS_STRIDE + ac + 8], gA + k0n + 8);
#pragma unroll
            for (int j = 0; j < 8; j++)
                async_ld_b128(&nW[tid * LDS_STRIDE + j * 8], gW + k0n + j * 8);
            WAIT_ASYNC_10();   // 10 newer ops in flight => chunk i has landed
        } else {
            WAIT_ASYNC_0();
        }
        __syncthreads();

#pragma unroll
        for (int kk = 0; kk < 64; kk += 32) {
            v16bf a[2], b[4];
#pragma unroll
            for (int tm = 0; tm < 2; tm++)
                a[tm] = *(const v16bf*)&curA[(wm * 32 + tm * 16 + lrow) * LDS_STRIDE + kk + lkh];
#pragma unroll
            for (int tn = 0; tn < 4; tn++)
                b[tn] = *(const v16bf*)&curW[(wn * 64 + tn * 16 + lrow) * LDS_STRIDE + kk + lkh];
#pragma unroll
            for (int tm = 0; tm < 2; tm++)
#pragma unroll
                for (int tn = 0; tn < 4; tn++)
                    acc[tm][tn] = __builtin_amdgcn_wmma_f32_16x16x32_bf16(
                        false, a[tm], false, b[tn], (short)0, acc[tm][tn], false, false);
        }
        __syncthreads();
    }

    // ---- epilogue: C/D layout — VGPR r: lanes 0-15 -> M=r, lanes 16-31 -> M=r+8 ----
    const int rbase = (lane >> 4) * 8;
    const int cl    = lane & 15;
#pragma unroll
    for (int tm = 0; tm < 2; tm++) {
#pragma unroll
        for (int tn = 0; tn < 4; tn++) {
#pragma unroll
            for (int r = 0; r < 8; r++) {
                int row = blockM + wm * 32 + tm * 16 + rbase + r;
                int col = blockN + wn * 64 + tn * 16 + cl;
                float v = acc[tm][tn][r];
                if (bias) v += bias[col];
                if (relu) v = fmaxf(v, 0.0f);
                if (Cf) Cf[(size_t)row * ldc + col] = v;
                if (Cb) Cb[(size_t)row * ldcb + col] = f2bf(v);
            }
        }
    }
}

// ---------------- LSTM pointwise ----------------
__global__ void lstm_cell(const float* __restrict__ gates,
                          float* __restrict__ c,
                          unsigned short* __restrict__ xh) {
    int idx = blockIdx.x * blockDim.x + threadIdx.x;   // over B*H
    if (idx >= Bsz * Hdim) return;
    int b = idx >> 9, j = idx & 511;
    const float* g = gates + (size_t)b * Ngate;
    float ig = sigmoidf_(g[j]);
    float fg = sigmoidf_(g[Hdim + j]);
    float gg = tanhf(g[2 * Hdim + j]);
    float og = sigmoidf_(g[3 * Hdim + j]);
    float cn = fg * c[idx] + ig * gg;
    float h  = og * tanhf(cn);
    c[idx] = cn;
    xh[(size_t)b * Kxh + Hdim + j] = f2bf(h);   // h half of concat buffer
}

// ---------------- softmax + Gumbel sample + outputs + embed gather ----------------
__global__ __launch_bounds__(256)
void sample_step(const float* __restrict__ logits,
                 const unsigned short* __restrict__ embb,
                 float* __restrict__ mask,
                 unsigned short* __restrict__ xh,
                 float* __restrict__ out, int t) {
    __shared__ float red[256];
    __shared__ float redv[256];
    __shared__ int   redi[256];
    __shared__ float s_m, s_lse, s_ent, s_mask;
    __shared__ int   s_act;

    const int b   = blockIdx.x;
    const int tid = threadIdx.x;
    const float* lrow = logits + (size_t)b * Vocab;

    if (tid == 0) s_mask = mask[b];

    // pass 1: max
    float m = -__builtin_inff();
    for (int v = tid; v < Vocab; v += 256) m = fmaxf(m, lrow[v]);
    red[tid] = m;
    __syncthreads();
    for (int s = 128; s > 0; s >>= 1) {
        if (tid < s) red[tid] = fmaxf(red[tid], red[tid + s]);
        __syncthreads();
    }
    if (tid == 0) s_m = red[0];
    __syncthreads();
    m = s_m;

    // pass 2: sum exp, sum exp*logit  -> lse, entropy
    float se = 0.0f, sl = 0.0f;
    for (int v = tid; v < Vocab; v += 256) {
        float l = lrow[v];
        float e = __expf(l - m);
        se += e;
        sl += e * l;
    }
    red[tid] = se;
    redv[tid] = sl;
    __syncthreads();
    for (int s = 128; s > 0; s >>= 1) {
        if (tid < s) { red[tid] += red[tid + s]; redv[tid] += redv[tid + s]; }
        __syncthreads();
    }
    if (tid == 0) {
        float seT = red[0];
        float lse = m + __logf(seT);
        s_lse = lse;
        s_ent = lse - redv[0] / seT;   // -sum p*(l-lse)
    }
    __syncthreads();
    const float lse = s_lse;

    // pass 3: Gumbel-max sampling (deterministic counter-based RNG)
    float bk = -__builtin_inff();
    int   bi = 0;
    unsigned long long base =
        ((unsigned long long)(t * Bsz + b)) * (unsigned long long)Vocab +
        0x2545F4914F6CDD1Dull * 42ull;
    for (int v = tid; v < Vocab; v += 256) {
        unsigned long long h = mix64(base + (unsigned long long)v);
        float u = ((float)(h >> 40) + 0.5f) * (1.0f / 16777216.0f);  // (0,1)
        float g = -__logf(-__logf(u));
        float key = lrow[v] + g;
        if (key > bk) { bk = key; bi = v; }
    }
    redv[tid] = bk;
    redi[tid] = bi;
    __syncthreads();
    for (int s = 128; s > 0; s >>= 1) {
        if (tid < s && redv[tid + s] > redv[tid]) {
            redv[tid] = redv[tid + s];
            redi[tid] = redi[tid + s];
        }
        __syncthreads();
    }
    if (tid == 0) s_act = redi[0];
    __syncthreads();

    const int   act = s_act;
    const float mo  = s_mask;

    float* lps   = out + (size_t)Bsz * Tlen;
    float* ents  = out + (size_t)2 * Bsz * Tlen;
    float* probs = out + (size_t)3 * Bsz * Tlen;

    if (tid == 0) {
        ((int*)out)[(size_t)b * Tlen + t] = act;                       // msgs (int32 bits)
        lps[(size_t)b * Tlen + t]  = (lrow[act] - lse) * mo;           // log-prob
        ents[(size_t)b * Tlen + t] = s_ent * mo;                       // entropy
        mask[b] = mo * ((act == 0) ? 0.0f : 1.0f);                     // eos update
    }

    float* pr = probs + ((size_t)b * Tlen + t) * Vocab;
    for (int v = tid; v < Vocab; v += 256)
        pr[v] = __expf(lrow[v] - lse) * mo;

    // x_next = emb[action]
    const unsigned short* er = embb + (size_t)act * Hdim;
    unsigned short* xr = xh + (size_t)b * Kxh;
    for (int j = tid; j < Hdim; j += 256) xr[j] = er[j];
}

// ---------------- workspace layout (bytes, 256-aligned) ----------------
#define OFF_XH     ((size_t)0)            // [B,1024] bf16   2 MB  ([x|h])
#define OFF_WCAT   ((size_t)2097152)      // [2048,1024] bf16 4 MB
#define OFF_OUTW   ((size_t)6291456)      // [8192,512] bf16  8 MB
#define OFF_EMB    ((size_t)14680064)     // [8192,512] bf16  8 MB
#define OFF_FC1W   ((size_t)23068672)     // [512,512] bf16   0.5 MB
#define OFF_TI     ((size_t)23592960)     // [1024,512] bf16  1 MB
#define OFF_BCAT   ((size_t)24641536)     // [2048] f32
#define OFF_C      ((size_t)24649728)     // [B,H] f32        2 MB
#define OFF_GATES  ((size_t)26746880)     // [B,2048] f32     8 MB
#define OFF_LOGITS ((size_t)35135488)     // [B,8192] f32     32 MB
#define OFF_MASK   ((size_t)68689920)     // [B] f32

extern "C" void kernel_launch(void* const* d_in, const int* in_sizes, int n_in,
                              void* d_out, int out_size, void* d_ws, size_t ws_size,
                              hipStream_t stream) {
    (void)in_sizes; (void)n_in; (void)out_size; (void)ws_size;

    const float* target_input = (const float*)d_in[0];
    const float* emb          = (const float*)d_in[1];
    const float* fc1_w        = (const float*)d_in[2];
    const float* fc1_b        = (const float*)d_in[3];
    const float* w_ih         = (const float*)d_in[4];
    const float* w_hh         = (const float*)d_in[5];
    const float* b_ih         = (const float*)d_in[6];
    const float* b_hh         = (const float*)d_in[7];
    const float* out_w        = (const float*)d_in[8];
    const float* out_b        = (const float*)d_in[9];

    char* ws = (char*)d_ws;
    unsigned short* xh    = (unsigned short*)(ws + OFF_XH);
    unsigned short* wcat  = (unsigned short*)(ws + OFF_WCAT);
    unsigned short* outwb = (unsigned short*)(ws + OFF_OUTW);
    unsigned short* embb  = (unsigned short*)(ws + OFF_EMB);
    unsigned short* fc1wb = (unsigned short*)(ws + OFF_FC1W);
    unsigned short* tib   = (unsigned short*)(ws + OFF_TI);
    float* bcat   = (float*)(ws + OFF_BCAT);
    float* cbuf   = (float*)(ws + OFF_C);
    float* gates  = (float*)(ws + OFF_GATES);
    float* logits = (float*)(ws + OFF_LOGITS);
    float* maskb  = (float*)(ws + OFF_MASK);

    const int TPB = 256;
    auto blocks = [](int n) { return (n + 255) / 256; };

    // ---- one-time (per launch) weight conversions & state init ----
    build_wcat<<<blocks(Ngate * Kxh), TPB, 0, stream>>>(w_ih, w_hh, wcat);
    build_bcat<<<blocks(Ngate), TPB, 0, stream>>>(b_ih, b_hh, bcat);
    cvt_f32_bf16<<<blocks(Vocab * Hdim), TPB, 0, stream>>>(out_w, outwb, Vocab * Hdim);
    cvt_f32_bf16<<<blocks(Vocab * Hdim), TPB, 0, stream>>>(emb, embb, Vocab * Hdim);
    cvt_f32_bf16<<<blocks(Hdim * Hdim), TPB, 0, stream>>>(fc1_w, fc1wb, Hdim * Hdim);
    cvt_f32_bf16<<<blocks(Bsz * Hdim), TPB, 0, stream>>>(target_input, tib, Bsz * Hdim);
    zero_f32<<<blocks(Bsz * Hdim), TPB, 0, stream>>>(cbuf, Bsz * Hdim);
    fill_f32<<<blocks(Bsz), TPB, 0, stream>>>(maskb, 1.0f, Bsz);
    zero_xh_x<<<blocks(Bsz * Hdim), TPB, 0, stream>>>(xh);

    // ---- h0 = ReLU(fc1(img)) -> bf16 into xh[:,512:1024] ----
    wmma_gemm_bf16<<<dim3(Bsz / 64, Hdim / 256), TPB, 0, stream>>>(
        tib, Hdim, fc1wb, Hdim, fc1_b,
        nullptr, 0, xh + Hdim, Kxh, Hdim, /*relu=*/1);

    // ---- T sequential decode steps ----
    for (int t = 0; t < Tlen; t++) {
        // gates = [x|h] @ [w_ih|w_hh]^T + (b_ih+b_hh)
        wmma_gemm_bf16<<<dim3(Bsz / 64, Ngate / 256), TPB, 0, stream>>>(
            xh, Kxh, wcat, Kxh, bcat,
            gates, Ngate, nullptr, 0, Kxh, 0);

        lstm_cell<<<blocks(Bsz * Hdim), TPB, 0, stream>>>(gates, cbuf, xh);

        // logits = h @ out_w^T + out_b
        wmma_gemm_bf16<<<dim3(Bsz / 64, Vocab / 256), TPB, 0, stream>>>(
            xh + Hdim, Kxh, outwb, Hdim, out_b,
            logits, Vocab, nullptr, 0, Hdim, 0);

        sample_step<<<Bsz, TPB, 0, stream>>>(logits, embb, maskb, xh,
                                             (float*)d_out, t);
    }
}
